// MultiWaveletCross_4123168604351
// MI455X (gfx1250) — compile-verified, matrix operand order checked by reading, not run on hardware
//
#include <hip/hip_runtime.h>
#include <math.h>

// Problem constants (from reference)
#define B_   16
#define N_   1024
#define C_   64
#define K_   8
#define CK_  512
#define M_ROWS (B_ * N_)          // 16384
#define BKC  (B_ * K_ * C_)       // 8192

typedef __attribute__((ext_vector_type(2))) float v2f;
typedef __attribute__((ext_vector_type(8))) float v8f;

#define HAVE_WMMA_F32 __has_builtin(__builtin_amdgcn_wmma_f32_16x16x4_f32)
#if !HAVE_WMMA_F32
typedef __attribute__((ext_vector_type(16))) __bf16 v16bf;
#endif

// ---------------------------------------------------------------------------
// GEMM: out[M x 512] = A[M x 512] @ W[512 x 512] + bias[512]
// 1 wave -> one 16x16 C tile; 4 waves/block share a 16x512 A strip in LDS.
// ---------------------------------------------------------------------------
__global__ __launch_bounds__(128) void gemm512_wmma(
    const float* __restrict__ A, const float* __restrict__ W,
    const float* __restrict__ bias, float* __restrict__ out, int M) {
  __shared__ float lA[16 * 516];            // stride 516 -> conflict-free frag reads
  const int tid  = threadIdx.x;
  const int wave = tid >> 5;
  const int lane = tid & 31;
  const int mtile = blockIdx.x * 16;
  const int ntile = blockIdx.y * 64 + wave * 16;
  (void)M;

  for (int i = tid; i < 16 * 512; i += 128) {
    int r = i >> 9, c = i & 511;
    lA[r * 516 + c] = A[(size_t)(mtile + r) * 512 + c];
  }
  __syncthreads();

  const int half = lane >> 4;               // 0: lanes 0-15, 1: lanes 16-31
  const int l16  = lane & 15;
  const int col  = ntile + l16;
  v8f acc = {};

#if HAVE_WMMA_F32
  // f32 path: V_WMMA_F32_16X16X4_F32, 128 MACs along K=512.
  for (int k = 0; k < 512; k += 4) {
    const int ka = k + half * 2;
    v2f a, b;
    a.x = lA[l16 * 516 + ka];
    a.y = lA[l16 * 516 + ka + 1];
    b.x = W[(size_t)ka * 512 + col];
    b.y = W[(size_t)(ka + 1) * 512 + col];
    acc = __builtin_amdgcn_wmma_f32_16x16x4_f32(
        false, a, false, b, (short)0, acc, false, false);
  }
#else
  // bf16 fallback: V_WMMA_F32_16X16X32_BF16, convert fragments on the fly.
  for (int k = 0; k < 512; k += 32) {
    const int kh = k + half * 8;
    v16bf a, b;
#pragma unroll
    for (int j = 0; j < 8; ++j) {
      a[j]     = (__bf16)lA[l16 * 516 + kh + j];
      a[8 + j] = (__bf16)lA[l16 * 516 + kh + 16 + j];
      b[j]     = (__bf16)W[(size_t)(kh + j) * 512 + col];
      b[8 + j] = (__bf16)W[(size_t)(kh + 16 + j) * 512 + col];
    }
    acc = __builtin_amdgcn_wmma_f32_16x16x32_bf16(
        false, a, false, b, (short)0, acc, false, false);
  }
#endif

  const float bv = bias[col];
#pragma unroll
  for (int j = 0; j < 8; ++j) {
    const int row = mtile + j + half * 8;   // C/D layout: vgpr j -> M=j (+8 hi lanes)
    out[(size_t)row * 512 + col] = acc[j] + bv;
  }
}

// ---------------------------------------------------------------------------
// Wavelet forward: x (B,n,C,K) -> d,s (B,n/2,C,K);  xa = [x_even | x_odd] (16)
// ---------------------------------------------------------------------------
__global__ void wavelet_fwd(const float* __restrict__ x,
                            const float* __restrict__ ec_d,
                            const float* __restrict__ ec_s,
                            float* __restrict__ d, float* __restrict__ s, int h) {
  const int total = B_ * h * C_ * K_;
  const int tid = blockIdx.x * blockDim.x + threadIdx.x;
  if (tid >= total) return;
  const int kk = tid & 7;
  int r = tid >> 3;
  const int c = r & 63; r >>= 6;
  const int t = r % h;
  const int b = r / h;
  const size_t be = (((size_t)b * (2 * h) + 2 * t) * C_ + c) * K_;
  const size_t bo = be + CK_;               // next sequence position
  float ad = 0.f, as = 0.f;
#pragma unroll
  for (int j = 0; j < 8; ++j) {
    const float xv = x[be + j];
    ad += xv * ec_d[j * K_ + kk];
    as += xv * ec_s[j * K_ + kk];
  }
#pragma unroll
  for (int j = 0; j < 8; ++j) {
    const float xv = x[bo + j];
    ad += xv * ec_d[(8 + j) * K_ + kk];
    as += xv * ec_s[(8 + j) * K_ + kk];
  }
  d[tid] = ad;
  s[tid] = as;
}

// ---------------------------------------------------------------------------
// Direct DFT of first m modes along sequence: x (B,L,C,K) -> X (B,K,C,m) cplx
// ---------------------------------------------------------------------------
__global__ void dft_modes(const float* __restrict__ x, float2* __restrict__ X,
                          int L, int m) {
  const int total = BKC * m;
  const int tid = blockIdx.x * blockDim.x + threadIdx.x;
  if (tid >= total) return;
  const int f = tid % m;                    // fastest -> wave-broadcast loads
  int r = tid / m;
  const int c = r & 63; r >>= 6;
  const int hh = r & 7;
  const int b = r >> 3;
  const float w = -6.2831853071795864f * (float)f / (float)L;
  const float* p = x + ((size_t)b * L) * CK_ + c * K_ + hh;
  float re = 0.f, im = 0.f;
  for (int t = 0; t < L; ++t) {
    const float v = p[(size_t)t * CK_];
    float sn, cs;
    __sincosf(w * (float)t, &sn, &cs);
    re += v * cs;                           // e^{-i theta}
    im += v * sn;
  }
  X[tid] = make_float2(re, im);
}

// ---------------------------------------------------------------------------
// Mode-space attention per (b,h): qk = tanh(Xq^T Xk over C), Y = qk @ Xk^T
// ---------------------------------------------------------------------------
__global__ __launch_bounds__(256) void mode_attn(const float2* __restrict__ Xq,
                                                 const float2* __restrict__ Xk,
                                                 float2* __restrict__ Y, int m) {
  __shared__ float2 sqk[256];
  const int bh = blockIdx.x;                // B*K = 128 blocks
  const float2* q = Xq + (size_t)bh * C_ * m;
  const float2* k = Xk + (size_t)bh * C_ * m;
  const int tid = threadIdx.x;
  const int mm = m * m;
  if (tid < mm) {
    const int x = tid / m, y = tid - x * m;
    float re = 0.f, im = 0.f;
    for (int c = 0; c < C_; ++c) {
      const float2 a = q[c * m + x], bb = k[c * m + y];
      re += a.x * bb.x - a.y * bb.y;
      im += a.x * bb.y + a.y * bb.x;
    }
    // complex tanh(a+ib) = (sinh2a + i sin2b) / (cosh2a + cos2b)
    float tr, ti;
    if (fabsf(re) > 20.f) { tr = (re > 0.f) ? 1.f : -1.f; ti = 0.f; }
    else {
      const float den = coshf(2.f * re) + cosf(2.f * im);
      tr = sinhf(2.f * re) / den;
      ti = sinf(2.f * im) / den;
    }
    sqk[tid] = make_float2(tr, ti);
  }
  __syncthreads();
  for (int e = tid; e < C_ * m; e += 256) {
    const int c = e / m, x = e - c * m;
    float re = 0.f, im = 0.f;
    for (int y = 0; y < m; ++y) {
      const float2 a = sqk[x * m + y], bb = k[c * m + y];
      re += a.x * bb.x - a.y * bb.y;
      im += a.x * bb.y + a.y * bb.x;
    }
    Y[(size_t)bh * C_ * m + e] = make_float2(re, im);
  }
}

// ---------------------------------------------------------------------------
// irfft of (Y1 [+ Y2]) low modes -> out (B,L,C,K), scaled by 1/(4096*L)
// ---------------------------------------------------------------------------
__global__ void irfft_out(const float2* __restrict__ Y1,
                          const float2* __restrict__ Y2,
                          float* __restrict__ out, int L, int m) {
  const int total = BKC * L;                // == B*L*C*K
  const int tid = blockIdx.x * blockDim.x + threadIdx.x;
  if (tid >= total) return;
  const int hh = tid & 7;
  int r = tid >> 3;
  const int c = r & 63; r >>= 6;
  const int t = r % L;
  const int b = r / L;
  const size_t sb = (((size_t)b * K_ + hh) * C_ + c) * m;
  const float w = 6.2831853071795864f * (float)t / (float)L;
  float acc = 0.f;
  for (int f = 0; f < m; ++f) {
    float2 X = Y1[sb + f];
    if (Y2) { const float2 X2 = Y2[sb + f]; X.x += X2.x; X.y += X2.y; }
    if (f == 0) acc += X.x;
    else {
      float sn, cs;
      __sincosf(w * (float)f, &sn, &cs);
      acc += 2.f * (X.x * cs - X.y * sn);
    }
  }
  out[tid] = acc * (1.0f / (4096.0f * (float)L));
}

// ---------------------------------------------------------------------------
// Reconstruction: fuse v+=Us, concat Ud, even/odd filters. vin L -> vout 2L
// ---------------------------------------------------------------------------
__global__ void recon_step(const float* __restrict__ vin,
                           const float* __restrict__ Us,
                           const float* __restrict__ Ud,
                           const float* __restrict__ rc_e,
                           const float* __restrict__ rc_o,
                           float* __restrict__ vout, int L) {
  const int total = B_ * 2 * L * C_ * K_;
  const int tid = blockIdx.x * blockDim.x + threadIdx.x;
  if (tid >= total) return;
  const int kk = tid & 7;
  int r = tid >> 3;
  const int c = r & 63; r >>= 6;
  const int tt = r % (2 * L);
  const int b = r / (2 * L);
  const int t = tt >> 1;
  const float* rc = (tt & 1) ? rc_o : rc_e;
  const size_t base = (((size_t)b * L + t) * C_ + c) * K_;
  float acc = 0.f;
#pragma unroll
  for (int j = 0; j < 8; ++j)
    acc += (vin[base + j] + Us[base + j]) * rc[j * K_ + kk];
#pragma unroll
  for (int j = 0; j < 8; ++j)
    acc += Ud[base + j] * rc[(8 + j) * K_ + kk];
  vout[tid] = acc;
}

// ---------------------------------------------------------------------------
static inline int cdiv(int a, int b) { return (a + b - 1) / b; }

extern "C" void kernel_launch(void* const* d_in, const int* in_sizes, int n_in,
                              void* d_out, int out_size, void* d_ws, size_t ws_size,
                              hipStream_t stream) {
  (void)in_sizes; (void)n_in; (void)out_size; (void)ws_size;
  const float* q_in  = (const float*)d_in[0];
  const float* k_in  = (const float*)d_in[1];
  // d_in[2] (v), d_in[7..8] (Lv_w/b) are provably unused by the reference math.
  const float* Lq_w  = (const float*)d_in[3];
  const float* Lq_b  = (const float*)d_in[4];
  const float* Lk_w  = (const float*)d_in[5];
  const float* Lk_b  = (const float*)d_in[6];
  const float* out_w = (const float*)d_in[9];
  const float* out_b = (const float*)d_in[10];
  const float* ec_s  = (const float*)d_in[11];
  const float* ec_d  = (const float*)d_in[12];
  const float* rc_e  = (const float*)d_in[13];
  const float* rc_o  = (const float*)d_in[14];
  float* out = (float*)d_out;

  // Workspace carve-out (floats)
  float* ws = (float*)d_ws;
  size_t o = 0;
  auto alloc = [&](size_t n) { float* p = ws + o; o += n; return p; };
  const size_t FULL = (size_t)B_ * N_ * CK_;       // 8,388,608
  const size_t HALF = FULL / 2;                    // 4,194,304
  float* QP    = alloc(FULL);                      // q projection; later recon ping
  float* KP    = alloc(FULL);                      // k projection; later recon pong
  float* sq[2] = { alloc(HALF), alloc(HALF) };
  float* dq    = alloc(HALF);
  float* sk[2] = { alloc(HALF), alloc(HALF) };
  float* dk    = alloc(HALF);
  float* UD    = alloc((size_t)BKC * 1023);
  float* US    = alloc((size_t)BKC * 1023);
  float2* SQd  = (float2*)alloc(2 * (size_t)BKC * 16);
  float2* SKd  = (float2*)alloc(2 * (size_t)BKC * 16);
  float2* SQs  = (float2*)alloc(2 * (size_t)BKC * 16);
  float2* SKs  = (float2*)alloc(2 * (size_t)BKC * 16);
  float2* Yd   = (float2*)alloc(2 * (size_t)BKC * 16);
  float2* Ys   = (float2*)alloc(2 * (size_t)BKC * 16);
  float* VZ    = alloc((size_t)BKC);               // coarsest v (always zero: m=0)

  const dim3 gblk(128), gdim(M_ROWS / 16, CK_ / 64);

  // Input projections (WMMA GEMMs)
  gemm512_wmma<<<gdim, gblk, 0, stream>>>(q_in, Lq_w, Lq_b, QP, M_ROWS);
  gemm512_wmma<<<gdim, gblk, 0, stream>>>(k_in, Lk_w, Lk_b, KP, M_ROWS);
  hipMemsetAsync(VZ, 0, (size_t)BKC * sizeof(float), stream);

  // Wavelet decomposition + Fourier cross-attention per level
  size_t off[10];
  const float* cq = QP;
  const float* ck = KP;
  size_t acc_off = 0;
  for (int i = 0; i < 10; ++i) {
    const int n = N_ >> i, h = n >> 1;
    int m = h >> 1; if (m > 16) m = 16;            // min(h//2, MODES)
    off[i] = acc_off; acc_off += (size_t)BKC * h;
    const int wv_total = B_ * h * CK_;
    wavelet_fwd<<<cdiv(wv_total, 256), 256, 0, stream>>>(cq, ec_d, ec_s, dq, sq[i & 1], h);
    wavelet_fwd<<<cdiv(wv_total, 256), 256, 0, stream>>>(ck, ec_d, ec_s, dk, sk[i & 1], h);
    if (m > 0) {
      const int ft = BKC * m;
      dft_modes<<<cdiv(ft, 256), 256, 0, stream>>>(dq,       SQd, h, m);
      dft_modes<<<cdiv(ft, 256), 256, 0, stream>>>(dk,       SKd, h, m);
      dft_modes<<<cdiv(ft, 256), 256, 0, stream>>>(sq[i & 1], SQs, h, m);
      dft_modes<<<cdiv(ft, 256), 256, 0, stream>>>(sk[i & 1], SKs, h, m);
      mode_attn<<<B_ * K_, 256, 0, stream>>>(SQd, SKd, Yd, m);
      mode_attn<<<B_ * K_, 256, 0, stream>>>(SQs, SKs, Ys, m);
    }
    const int it = BKC * h;
    irfft_out<<<cdiv(it, 256), 256, 0, stream>>>(Yd, Ys,      UD + off[i], h, m); // Ud = fca(d)+fca(s)
    irfft_out<<<cdiv(it, 256), 256, 0, stream>>>(Yd, nullptr, US + off[i], h, m); // Us = fca(d)
    cq = sq[i & 1];
    ck = sk[i & 1];
  }

  // Reconstruction: coarsest v is fca at L=1 -> zeros (VZ). Ping-pong QP/KP.
  const float* vcur = VZ;
  for (int i = 9; i >= 0; --i) {
    const int L = 512 >> i;                        // length of Us[i]/Ud[i]
    float* vnext = ((9 - i) & 1) ? KP : QP;
    const int total = B_ * 2 * L * CK_;
    recon_step<<<cdiv(total, 256), 256, 0, stream>>>(vcur, US + off[i], UD + off[i],
                                                     rc_e, rc_o, vnext, L);
    vcur = vnext;
  }

  // Output projection (WMMA GEMM) straight into d_out
  gemm512_wmma<<<gdim, gblk, 0, stream>>>(vcur, out_w, out_b, out, M_ROWS);
}
